// BinaryTreeLSTM_65017214926874
// MI455X (gfx1250) — matrix operand order repaired and checked by reading, not compiled, not run on hardware
//
#include <hip/hip_runtime.h>
#include <math.h>

typedef __attribute__((ext_vector_type(16))) _Float16 v16h;
typedef __attribute__((ext_vector_type(8)))  _Float16 v8h;
typedef __attribute__((ext_vector_type(8)))  float    v8f;

static __device__ __forceinline__ float sigf(float x) {
    return 1.0f / (1.0f + __expf(-x));
}

// ---------------------------------------------------------------------------
// C(M x N) = A(M x K, f16 row-major) @ W(N x K, f16 row-major)^T  (f32 accum)
// One wave per block; each wave computes a 16x64 tile via 4 WMMAs per K-step.
// K must be a multiple of 32; N a multiple of 64; M is bounds-guarded.
// ---------------------------------------------------------------------------
__global__ __launch_bounds__(32) void wmma_gemm_nt(
    const _Float16* __restrict__ A, const _Float16* __restrict__ W,
    float* __restrict__ C, int M, int N, int K)
{
    const int lane = threadIdx.x;
    const int l16  = lane & 15;
    const int half = lane >> 4;           // 0 or 1
    const int mbase = blockIdx.x << 4;
    const int nbase = blockIdx.y << 6;

    int arow = mbase + l16;
    if (arow >= M) arow = M - 1;          // clamp: OOB rows never stored
    const _Float16* ap = A + (size_t)arow * K + half * 16;
    const _Float16* wp = W + (size_t)(nbase + l16) * K + half * 16;
    const size_t wstride = (size_t)16 * K;

    v8f acc0 = {}, acc1 = {}, acc2 = {}, acc3 = {};

    for (int k0 = 0; k0 < K; k0 += 32) {
        // A fragment: 16x32 f16. lane holds row l%16, 16 contiguous K values from half*16.
        v8h alo = *(const v8h*)(ap + k0);
        v8h ahi = *(const v8h*)(ap + k0 + 8);
        v16h a;
#pragma unroll
        for (int e = 0; e < 8; ++e) { a[e] = alo[e]; a[e + 8] = ahi[e]; }

        // B fragments: 32x16. lane holds col N=l%16, 16 contiguous K values.
        v16h b0 = *(const v16h*)(wp + k0);
        v16h b1 = *(const v16h*)(wp + k0 + wstride);
        v16h b2 = *(const v16h*)(wp + k0 + 2 * wstride);
        v16h b3 = *(const v16h*)(wp + k0 + 3 * wstride);

        acc0 = __builtin_amdgcn_wmma_f32_16x16x32_f16(false, a, false, b0, (short)0, acc0, false, false);
        acc1 = __builtin_amdgcn_wmma_f32_16x16x32_f16(false, a, false, b1, (short)0, acc1, false, false);
        acc2 = __builtin_amdgcn_wmma_f32_16x16x32_f16(false, a, false, b2, (short)0, acc2, false, false);
        acc3 = __builtin_amdgcn_wmma_f32_16x16x32_f16(false, a, false, b3, (short)0, acc3, false, false);
    }

    // D layout: element v -> row mbase + v + half*8, col nbase + l16
    const int row0 = mbase + half * 8;
    const int col  = nbase + l16;
    float* cp = C + (size_t)row0 * N + col;
#pragma unroll
    for (int v = 0; v < 8; ++v) {
        if (row0 + v < M) {
            cp[(size_t)v * N +  0] = acc0[v];
            cp[(size_t)v * N + 16] = acc1[v];
            cp[(size_t)v * N + 32] = acc2[v];
            cp[(size_t)v * N + 48] = acc3[v];
        }
    }
}

// ---- f32 -> f16 with K-padding (embs 300->320, Wx 300->320) ---------------
__global__ __launch_bounds__(256) void cvt_pad_f16(
    const float* __restrict__ src, _Float16* __restrict__ dst,
    int rows, int kin, int kout)
{
    int i = blockIdx.x * 256 + threadIdx.x;
    if (i >= rows * kout) return;
    int r = i / kout, k = i - r * kout;
    dst[i] = (k < kin) ? (_Float16)src[(size_t)r * kin + k] : (_Float16)0.0f;
}

// ---- Pack [Wl | Wr] into one (NGLR x 512) f16 weight matrix ---------------
__global__ __launch_bounds__(256) void cvt_wlr(
    const float* __restrict__ Wl, const float* __restrict__ Wr,
    _Float16* __restrict__ dst, int N, int K)
{
    int i = blockIdx.x * 256 + threadIdx.x;
    if (i >= N * 2 * K) return;
    int r = i / (2 * K), k = i - r * 2 * K;
    float v = (k < K) ? Wl[(size_t)r * K + k] : Wr[(size_t)r * K + (k - K)];
    dst[i] = (_Float16)v;
}

// ---- pad_xg = Wx @ emb_table[-1] + bx  (1024 dots of length 300, f32) -----
__global__ __launch_bounds__(256) void pad_proj(
    const float* __restrict__ Wx, const float* __restrict__ bx,
    const float* __restrict__ emb_row, float* __restrict__ pad, int NG, int K)
{
    int j = blockIdx.x * 256 + threadIdx.x;
    if (j >= NG) return;
    float s = bx[j];
    const float* w = Wx + (size_t)j * K;
    for (int k = 0; k < K; ++k) s += w[k] * emb_row[k];
    pad[j] = s;
}

// ---- Leaf gates: children are zero -> c = i*u, h = o*tanh(c) --------------
__global__ __launch_bounds__(256) void leaf_pointwise(
    const float* __restrict__ g, const float* __restrict__ bx,
    float* __restrict__ c_out, _Float16* __restrict__ lhrh_out)
{
    const int b = blockIdx.x;
    const int d = threadIdx.x;                 // 0..255
    const float* gr = g + (size_t)b * 1024;
    float u = tanhf(gr[d]       + bx[d]);
    float i = sigf (gr[256 + d] + bx[256 + d]);
    float c = i * u;
    float o = sigf (gr[768 + d] + bx[768 + d]);
    float h = o * tanhf(c);
    c_out[(size_t)b * 256 + d] = c;
    // pack h into parent row (b>>1), left/right half selected by b&1
    lhrh_out[(size_t)(b >> 1) * 512 + (b & 1) * 256 + d] = (_Float16)h;
}

// ---- Internal-node gates: g = lg+rg (B x 1280), pad_xg constant -----------
__global__ __launch_bounds__(256) void compose_pointwise(
    const float* __restrict__ g, const float* __restrict__ pad,
    const float* __restrict__ c_prev, float* __restrict__ c_out,
    _Float16* __restrict__ lhrh_out, float* __restrict__ out_final)
{
    const int b = blockIdx.x;
    const int d = threadIdx.x;
    const float* gr = g + (size_t)b * 1280;
    float u  = tanhf(pad[d]        + gr[d]);
    float ig = sigf (pad[256 + d]  + gr[256 + d]);
    float lf = sigf (pad[512 + d]  + gr[512 + d]);
    float rf = sigf (pad[512 + d]  + gr[768 + d]);   // rf shares pad gate 2
    float lc = c_prev[(size_t)(2 * b)     * 256 + d];
    float rc = c_prev[(size_t)(2 * b + 1) * 256 + d];
    float c  = ig * u + lf * lc + rf * rc;
    float o  = sigf (pad[768 + d] + gr[1024 + d]);
    float h  = o * tanhf(c);
    c_out[(size_t)b * 256 + d] = c;
    if (lhrh_out)
        lhrh_out[(size_t)(b >> 1) * 512 + (b & 1) * 256 + d] = (_Float16)h;
    if (out_final) { out_final[d] = c; out_final[256 + d] = h; }
}

extern "C" void kernel_launch(void* const* d_in, const int* in_sizes, int n_in,
                              void* d_out, int out_size, void* d_ws, size_t ws_size,
                              hipStream_t stream)
{
    const float* embs = (const float*)d_in[0];  // (NL, 300)
    const float* Wx   = (const float*)d_in[1];  // (1024, 300)
    const float* bx   = (const float*)d_in[2];  // (1024,)
    const float* Wl   = (const float*)d_in[3];  // (1280, 256)
    const float* Wr   = (const float*)d_in[4];  // (1280, 256)
    const float* et   = (const float*)d_in[5];  // (NW, 300)
    float* out = (float*)d_out;                 // c(256) ++ h(256)

    const int IN  = 300, KIN = 320, M = 256;
    const int NL   = in_sizes[0] / IN;          // 8192
    const int NGX  = in_sizes[2];               // 1024
    const int NGLR = in_sizes[3] / M;           // 1280
    const int NW   = in_sizes[5] / IN;          // 50000

    char* base = (char*)d_ws;
    size_t off = 0;
    auto take = [&](size_t bytes) -> char* {
        char* p = base + off;
        off += (bytes + 255) & ~(size_t)255;
        return p;
    };

    _Float16* embs16 = (_Float16*)take((size_t)NL * KIN * 2);
    _Float16* Wx16   = (_Float16*)take((size_t)NGX * KIN * 2);
    _Float16* Wlr16  = (_Float16*)take((size_t)NGLR * 2 * M * 2);
    float*    padxg  = (float*)take((size_t)NGX * 4);
    size_t gb1 = (size_t)NL * NGX * 4, gb2 = (size_t)(NL / 2) * NGLR * 4;
    float*    g      = (float*)take(gb1 > gb2 ? gb1 : gb2);
    float*    cA     = (float*)take((size_t)NL * M * 4);
    float*    cB     = (float*)take((size_t)(NL / 2) * M * 4);
    _Float16* lA     = (_Float16*)take((size_t)(NL / 2) * 2 * M * 2);
    _Float16* lB     = (_Float16*)take((size_t)(NL / 4) * 2 * M * 2);

    // ---- precompute: f16 conversions + pad_xg ----
    {
        int n = NL * KIN;
        cvt_pad_f16<<<(n + 255) / 256, 256, 0, stream>>>(embs, embs16, NL, IN, KIN);
        n = NGX * KIN;
        cvt_pad_f16<<<(n + 255) / 256, 256, 0, stream>>>(Wx, Wx16, NGX, IN, KIN);
        n = NGLR * 2 * M;
        cvt_wlr<<<(n + 255) / 256, 256, 0, stream>>>(Wl, Wr, Wlr16, NGLR, M);
    }
    pad_proj<<<(NGX + 255) / 256, 256, 0, stream>>>(
        Wx, bx, et + (size_t)(NW - 1) * IN, padxg, NGX, IN);

    // ---- leaf stage: (NL x 1024) = embs16 @ Wx16^T, then gates ----
    {
        dim3 gl(NL / 16, NGX / 64);
        wmma_gemm_nt<<<gl, 32, 0, stream>>>(embs16, Wx16, g, NL, NGX, KIN);
        leaf_pointwise<<<NL, 256, 0, stream>>>(g, bx, cA, lA);
    }

    // ---- 13 tree levels: one fused [lh|rh] @ [Wl|Wr]^T GEMM per level ----
    float*    c_prev = cA;
    _Float16* lh     = lA;
    float*    c_nexts[2]  = {cB, cA};
    _Float16* lh_nexts[2] = {lB, lA};
    int ping = 0;
    for (int B = NL / 2; B >= 1; B >>= 1) {
        dim3 gg((B + 15) / 16, NGLR / 64);
        wmma_gemm_nt<<<gg, 32, 0, stream>>>(lh, Wlr16, g, B, NGLR, 2 * M);
        float*    c_next  = c_nexts[ping];
        _Float16* lh_next = (B > 1) ? lh_nexts[ping] : nullptr;
        compose_pointwise<<<B, 256, 0, stream>>>(
            g, padxg, c_prev, c_next, lh_next, (B == 1) ? out : nullptr);
        c_prev = c_next;
        lh = lh_nexts[ping];
        ping ^= 1;
    }
}
